// LoRAAttention_86595130622104
// MI455X (gfx1250) — matrix operand order repaired and checked by reading, not compiled, not run on hardware
//
#include <hip/hip_runtime.h>
#include <hip/hip_bf16.h>

// ---------------------------------------------------------------------------
// MI455X (gfx1250) fused attention: QKV GEMM -> RoPE -> flash attn -> O GEMM
// All matmuls on v_wmma_f32_16x16x32_bf16 (wave32, 16x16 tiles, f32 accum).
// ---------------------------------------------------------------------------

typedef __attribute__((ext_vector_type(16))) __bf16 v16bf;
typedef __attribute__((ext_vector_type(8)))  float  v8f;

#define BB 2
#define SS 2048
#define DD 4096
#define HH 32
#define HD 128

union Frag { unsigned int u[8]; v16bf v; };

__device__ __forceinline__ unsigned short f2bf(float f) {
    unsigned int u = __builtin_bit_cast(unsigned int, f);
    u += 0x7fffu + ((u >> 16) & 1u);         // round-to-nearest-even
    return (unsigned short)(u >> 16);
}
__device__ __forceinline__ float bf2f(unsigned short h) {
    unsigned int u = ((unsigned int)h) << 16;
    return __builtin_bit_cast(float, u);
}
__device__ __forceinline__ v8f vzero() {
    v8f r;
#pragma unroll
    for (int i = 0; i < 8; ++i) r[i] = 0.f;
    return r;
}

// ---------------------------------------------------------------------------
// Generic 128x128x32-tile WMMA GEMM: C[M,N] = A[M,K] * B[K,N]
// 256 threads = 8 waves; each wave: 64x32 (4x2 accumulator fragments).
// A source fp32 or bf16 (template), B source fp32; C out fp32 or bf16.
// ---------------------------------------------------------------------------
template <int ABF16, int OBF16>
__global__ __launch_bounds__(256) void gemm_wmma_128(
    const void* __restrict__ Ap, const float* __restrict__ Bp,
    void* __restrict__ Cp, int M, int N, int K, int lda, int ldb, int ldc)
{
    __shared__ unsigned short As[128][36];   // [m][k]  (row 72B, 8B aligned)
    __shared__ unsigned short Bs[128][36];   // [n][k]  (transposed on store)

    const int tid  = threadIdx.x;
    const int lane = tid & 31;
    const int wave = tid >> 5;
    const int l16  = lane & 15;
    const int half = lane >> 4;
    const int wm   = (wave >> 2) * 64;
    const int wn   = (wave & 3) * 32;
    const int m0   = blockIdx.y * 128;
    const int n0   = blockIdx.x * 128;

    const float*          Af = (const float*)Ap;
    const unsigned short* Ah = (const unsigned short*)Ap;

    v8f acc[4][2];
#pragma unroll
    for (int mt = 0; mt < 4; ++mt)
#pragma unroll
        for (int nt = 0; nt < 2; ++nt) acc[mt][nt] = vzero();

    for (int k0 = 0; k0 < K; k0 += 32) {
        // ---- stage A tile (128x32) into LDS as bf16 ----
        {
            const int arow = tid >> 3;
            const int acol = (tid & 7) * 4;
#pragma unroll
            for (int r = 0; r < 4; ++r) {
                const int row = arow + r * 32;
                union { uint2 d2; unsigned short h[4]; } t;
                if (ABF16) {
                    t.d2 = *(const uint2*)(Ah + (size_t)(m0 + row) * lda + k0 + acol);
                } else {
                    float4 d = *(const float4*)(Af + (size_t)(m0 + row) * lda + k0 + acol);
                    t.h[0] = f2bf(d.x); t.h[1] = f2bf(d.y);
                    t.h[2] = f2bf(d.z); t.h[3] = f2bf(d.w);
                }
                *(uint2*)&As[row][acol] = t.d2;
            }
        }
        // ---- stage B tile (32x128) transposed into LDS as bf16 ----
        {
            const int brow = tid >> 5;
            const int bcol = (tid & 31) * 4;
#pragma unroll
            for (int r = 0; r < 4; ++r) {
                const int kk = brow + r * 8;
                float4 d = *(const float4*)(Bp + (size_t)(k0 + kk) * ldb + n0 + bcol);
                Bs[bcol + 0][kk] = f2bf(d.x);
                Bs[bcol + 1][kk] = f2bf(d.y);
                Bs[bcol + 2][kk] = f2bf(d.z);
                Bs[bcol + 3][kk] = f2bf(d.w);
            }
        }
        if (k0 + 32 < K) {   // hint next A tile toward L2 (global_prefetch_b8)
            const char* pa = (const char*)Ap +
                ((size_t)(m0 + (tid >> 3)) * lda + k0 + 32) * (ABF16 ? 2 : 4);
            __builtin_prefetch(pa, 0, 0);
        }
        __syncthreads();

        // ---- fragments + WMMA ----
        Frag a[4], b[2];
#pragma unroll
        for (int mt = 0; mt < 4; ++mt) {
            const unsigned short* rp = &As[wm + mt * 16 + l16][0];
#pragma unroll
            for (int p = 0; p < 8; ++p)
                a[mt].u[p] = *(const unsigned int*)(rp + ((p >> 2) << 4) + (half << 3) + ((p & 3) << 1));
        }
#pragma unroll
        for (int nt = 0; nt < 2; ++nt) {
            const unsigned short* rp = &Bs[wn + nt * 16 + l16][0];
#pragma unroll
            for (int p = 0; p < 8; ++p)
                b[nt].u[p] = *(const unsigned int*)(rp + ((p >> 2) << 4) + (half << 3) + ((p & 3) << 1));
        }
#pragma unroll
        for (int mt = 0; mt < 4; ++mt)
#pragma unroll
            for (int nt = 0; nt < 2; ++nt)
                acc[mt][nt] = __builtin_amdgcn_wmma_f32_16x16x32_bf16(
                    false, a[mt].v, false, b[nt].v, (short)0, acc[mt][nt], false, false);
        __syncthreads();
    }

    // ---- epilogue ----
#pragma unroll
    for (int mt = 0; mt < 4; ++mt)
#pragma unroll
        for (int nt = 0; nt < 2; ++nt)
#pragma unroll
            for (int i = 0; i < 8; ++i) {
                const int row = m0 + wm + mt * 16 + (half << 3) + i;
                const int col = n0 + wn + nt * 16 + l16;
                if (OBF16)
                    ((unsigned short*)Cp)[(size_t)row * ldc + col] = f2bf(acc[mt][nt][i]);
                else
                    ((float*)Cp)[(size_t)row * ldc + col] = acc[mt][nt][i];
            }
}

// ---------------------------------------------------------------------------
// RoPE applied in place on bf16 q,k inside the qkv buffer.
// Deterministic across replays: QKV GEMM fully rewrites qkv before this runs.
// ---------------------------------------------------------------------------
__global__ __launch_bounds__(256) void rope_kernel(
    const int* __restrict__ positions, unsigned short* __restrict__ qkv)
{
    const int idx = blockIdx.x * 256 + threadIdx.x;
    const int d   = idx & 63;
    const int h   = (idx >> 6) & 31;
    const int row = idx >> 11;                    // b*S + s
    const float inv = __expf(-(float)d * (9.210340371976184f / 64.0f)); // 10000^(-d/64)
    const float ang = (float)positions[row] * inv;
    float sn, cs;
    __sincosf(ang, &sn, &cs);
    unsigned short* q = qkv + (size_t)row * (3 * DD) + h * HD;
    unsigned short* k = q + DD;
    float q1 = bf2f(q[d]), q2 = bf2f(q[d + 64]);
    q[d]      = f2bf(q1 * cs - q2 * sn);
    q[d + 64] = f2bf(q2 * cs + q1 * sn);
    float k1 = bf2f(k[d]), k2 = bf2f(k[d + 64]);
    k[d]      = f2bf(k1 * cs - k2 * sn);
    k[d + 64] = f2bf(k2 * cs + k1 * sn);
}

// ---------------------------------------------------------------------------
// Flash attention: one block = (b, h, 128 q rows). 8 waves x 16 q rows each.
// Q fragments in registers; K / V^T tiles (64 kv rows) staged in LDS.
// Online softmax with half-wave (16-lane) xor reductions matching the
// 16x16 C-fragment row layout.
// ---------------------------------------------------------------------------
__global__ __launch_bounds__(256) void flash_kernel(
    const unsigned short* __restrict__ qkv, unsigned short* __restrict__ attn)
{
    __shared__ unsigned short Ks[64][136];   // [kv][hd]
    __shared__ unsigned short Vt[128][72];   // [hd][kv] (transposed)
    __shared__ unsigned short Ps[8][16][72]; // per-wave P tile [m][kv]

    const int tid  = threadIdx.x;
    const int lane = tid & 31;
    const int wave = tid >> 5;
    const int l16  = lane & 15;
    const int half = lane >> 4;
    const int b    = blockIdx.z;
    const int h    = blockIdx.y;
    const int q0   = blockIdx.x * 128;
    const int ld   = 3 * DD;

    // ---- load Q fragments (16 rows x 128 hd = 4 A-fragments) ----
    Frag qf[4];
    {
        const size_t base = (size_t)(b * SS + q0 + wave * 16 + l16) * ld + h * HD;
#pragma unroll
        for (int f = 0; f < 4; ++f)
#pragma unroll
            for (int p = 0; p < 8; ++p)
                qf[f].u[p] = *(const unsigned int*)(qkv + base + f * 32 +
                             ((p >> 2) << 4) + (half << 3) + ((p & 3) << 1));
    }

    v8f o[8];
#pragma unroll
    for (int i = 0; i < 8; ++i) o[i] = vzero();
    float Mi[8], Li[8];
#pragma unroll
    for (int i = 0; i < 8; ++i) { Mi[i] = -1e30f; Li[i] = 0.f; }

    const float scale = 0.08838834764831845f;   // 1/sqrt(128)
    const int wq0 = q0 + wave * 16;

    for (int kv0 = 0; kv0 < q0 + 128; kv0 += 64) {
        // ---- cooperative stage of K tile and transposed V tile ----
#pragma unroll
        for (int r = 0; r < 4; ++r) {
            const int kr = (tid >> 4) + r * 16;
            const int c8 = (tid & 15) * 8;
            const unsigned short* kp =
                qkv + (size_t)(b * SS + kv0 + kr) * ld + DD + h * HD + c8;
            *(uint4*)&Ks[kr][c8] = *(const uint4*)kp;
            const unsigned short* vp = kp + DD;
#pragma unroll
            for (int j = 0; j < 8; ++j) Vt[c8 + j][kr] = vp[j];
        }
        __syncthreads();

        if (kv0 <= wq0 + 15) {     // wave-uniform causal skip (EXEC stays full)
            // ---- scores = Q x K^T : 4 n-subtiles x 4 K-chunks ----
            v8f sc[4];
#pragma unroll
            for (int nt = 0; nt < 4; ++nt) sc[nt] = vzero();
#pragma unroll
            for (int nt = 0; nt < 4; ++nt) {
                const unsigned short* rp = &Ks[nt * 16 + l16][0];
#pragma unroll
                for (int kc = 0; kc < 4; ++kc) {
                    Frag bk;
#pragma unroll
                    for (int p = 0; p < 8; ++p)
                        bk.u[p] = *(const unsigned int*)(rp + kc * 32 + ((p >> 2) << 4) + (half << 3) + ((p & 3) << 1));
                    sc[nt] = __builtin_amdgcn_wmma_f32_16x16x32_bf16(
                        false, qf[kc].v, false, bk.v, (short)0, sc[nt], false, false);
                }
            }
            // ---- online softmax (rows live in 16-lane halves) ----
#pragma unroll
            for (int i = 0; i < 8; ++i) {
                const int qr = wq0 + (half << 3) + i;
                float mloc = -1e30f;
#pragma unroll
                for (int nt = 0; nt < 4; ++nt) {
                    const int kvr = kv0 + nt * 16 + l16;
                    float s = sc[nt][i] * scale;
                    s = (kvr > qr) ? -1e30f : s;
                    sc[nt][i] = s;
                    mloc = fmaxf(mloc, s);
                }
#pragma unroll
                for (int off = 1; off < 16; off <<= 1)
                    mloc = fmaxf(mloc, __shfl_xor(mloc, off, 32));
                const float nm    = fmaxf(Mi[i], mloc);
                const float alpha = __expf(Mi[i] - nm);
                Mi[i] = nm;
                float rs = 0.f;
#pragma unroll
                for (int nt = 0; nt < 4; ++nt) {
                    const float pv = __expf(sc[nt][i] - nm);
                    rs += pv;
                    Ps[wave][(half << 3) + i][nt * 16 + l16] = f2bf(pv);
                }
#pragma unroll
                for (int off = 1; off < 16; off <<= 1)
                    rs += __shfl_xor(rs, off, 32);
                Li[i] = Li[i] * alpha + rs;
#pragma unroll
                for (int ht = 0; ht < 8; ++ht) o[ht][i] *= alpha;
            }
            // ---- O += P x V : 2 K-chunks x 8 hd-subtiles ----
#pragma unroll
            for (int kc2 = 0; kc2 < 2; ++kc2) {
                Frag pa;
                const unsigned short* pp = &Ps[wave][l16][0];
#pragma unroll
                for (int p = 0; p < 8; ++p)
                    pa.u[p] = *(const unsigned int*)(pp + kc2 * 32 +
                              ((p >> 2) << 4) + (half << 3) + ((p & 3) << 1));
#pragma unroll
                for (int ht = 0; ht < 8; ++ht) {
                    Frag bv;
                    const unsigned short* vr = &Vt[ht * 16 + l16][0];
#pragma unroll
                    for (int p = 0; p < 8; ++p)
                        bv.u[p] = *(const unsigned int*)(vr + kc2 * 32 + ((p >> 2) << 4) + (half << 3) + ((p & 3) << 1));
                    o[ht] = __builtin_amdgcn_wmma_f32_16x16x32_bf16(
                        false, pa.v, false, bv.v, (short)0, o[ht], false, false);
                }
            }
        }
        __syncthreads();
    }

    // ---- normalize and write bf16 attention output (B,S,D) ----
#pragma unroll
    for (int ht = 0; ht < 8; ++ht)
#pragma unroll
        for (int i = 0; i < 8; ++i) {
            const int m = (half << 3) + i;
            const size_t row = (size_t)(b * SS + q0 + wave * 16 + m);
            attn[row * DD + h * HD + ht * 16 + l16] = f2bf(o[ht][i] / Li[i]);
        }
}

// ---------------------------------------------------------------------------
extern "C" void kernel_launch(void* const* d_in, const int* in_sizes, int n_in,
                              void* d_out, int out_size, void* d_ws, size_t ws_size,
                              hipStream_t stream)
{
    (void)in_sizes; (void)n_in; (void)out_size; (void)ws_size;
    const int*   positions = (const int*)d_in[0];
    const float* hidden    = (const float*)d_in[1];
    const float* Wqkv      = (const float*)d_in[2];
    const float* Wo        = (const float*)d_in[3];
    float*       out       = (float*)d_out;

    unsigned short* qkv_ws  = (unsigned short*)d_ws;                 // B*S*3D bf16 (96 MB)
    unsigned short* attn_ws = qkv_ws + (size_t)BB * SS * 3 * DD;     // B*S*D  bf16 (32 MB)

    const int M = BB * SS;   // 4096

    // 1) QKV projection: (4096 x 4096) x (4096 x 12288) -> bf16 qkv
    gemm_wmma_128<0, 1><<<dim3((3 * DD) / 128, M / 128), 256, 0, stream>>>(
        hidden, Wqkv, qkv_ws, M, 3 * DD, DD, DD, 3 * DD, 3 * DD);

    // 2) RoPE on q,k in place
    rope_kernel<<<(BB * SS * HH * (HD / 2)) / 256, 256, 0, stream>>>(positions, qkv_ws);

    // 3) Causal flash attention -> bf16 attn
    flash_kernel<<<dim3(SS / 128, HH, BB), 256, 0, stream>>>(qkv_ws, attn_ws);

    // 4) Output projection: (4096 x 4096) x (4096 x 4096) -> fp32 out
    gemm_wmma_128<1, 0><<<dim3(DD / 128, M / 128), 256, 0, stream>>>(
        attn_ws, Wo, out, M, DD, DD, DD, DD, DD);
}
